// MambaLiteHead_42683384988236
// MI455X (gfx1250) — matrix operand order repaired and checked by reading, not compiled, not run on hardware
//
#include <hip/hip_runtime.h>
#include <hip/hip_bf16.h>
#include <cstdint>

// ---------------------------------------------------------------------------
// MambaLiteHead for MI455X (gfx1250, wave32, WMMA).
//   B=8, T=8192, D=S=128, chunk=64.
//   4 GEMMs done with v_wmma_f32_16x16x32_bf16 (f32 accumulate).
//   Scan parallelized per 64-token chunk + inter-chunk prefix pass.
//   x tile staged via TENSOR_LOAD_TO_LDS (TDM); a/u/C tiles via
//   GLOBAL_LOAD_ASYNC_TO_LDS_B128. Epilogue uses v_exp_f32/v_log_f32.
// ---------------------------------------------------------------------------

#define DEV __device__ __forceinline__

typedef __bf16 bf16;
typedef __attribute__((ext_vector_type(16))) __bf16 v16bf;
typedef __attribute__((ext_vector_type(8)))  __bf16 bf16x8;
typedef __attribute__((ext_vector_type(8)))  float  v8f;
typedef __attribute__((ext_vector_type(4)))  float  f32x4;
typedef int v4i __attribute__((vector_size(16)));
typedef __attribute__((ext_vector_type(4))) unsigned int u32x4;
typedef __attribute__((ext_vector_type(8))) int          i32x8;
typedef __attribute__((ext_vector_type(4))) int          i32x4;

#define AS1 __attribute__((address_space(1)))
#define AS3 __attribute__((address_space(3)))

#define XPAD 136   // bf16 row stride for LDS tiles: 272B = 16B aligned, conflict-free A loads

// ---- CDNA5 async global->LDS copy (16B) ----
DEV void async_copy16(const void* gsrc, void* ldst) {
#if __has_builtin(__builtin_amdgcn_global_load_async_to_lds_b128)
  __builtin_amdgcn_global_load_async_to_lds_b128((AS1 v4i*)gsrc, (AS3 v4i*)ldst, 0, 0);
#else
  *(f32x4*)ldst = *(const f32x4*)gsrc;   // synchronous fallback
#endif
}

DEV void wait_async() {
#if __has_builtin(__builtin_amdgcn_s_wait_asynccnt)
  __builtin_amdgcn_s_wait_asynccnt(0);
#elif __has_builtin(__builtin_amdgcn_global_load_async_to_lds_b128)
  asm volatile("s_wait_asynccnt 0" ::: "memory");
#endif
}

// ---- hardware transcendentals (v_exp_f32 / v_log_f32, base-2) ----
DEV float fast_exp2(float x) {
#if __has_builtin(__builtin_amdgcn_exp2f)
  return __builtin_amdgcn_exp2f(x);
#else
  return exp2f(x);
#endif
}
DEV float fast_log2(float x) {
#if __has_builtin(__builtin_amdgcn_logf)
  return __builtin_amdgcn_logf(x);
#else
  return log2f(x);
#endif
}
#define LOG2E 1.4426950408889634f
#define LN2   0.6931471805599453f

// ---- WMMA fragment helpers (layouts per CDNA5 ISA 7.12.2, wave32) ----

// A-matrix 16x32 bf16: lane<16 -> K=[k0..k0+7],[k0+16..k0+23] with k0=kb*32;
// lane>=16 same +8. Two contiguous 16B LDS loads per lane.
DEV v16bf load_a_frag(const bf16* base, int stride, int m0, int kb, int lane) {
  const int row = m0 + (lane & 15);
  const int k0  = kb * 32 + ((lane >> 4) << 3);
  const bf16* p = base + row * stride + k0;
  bf16x8 lo = *(const bf16x8*)p;
  bf16x8 hi = *(const bf16x8*)(p + 16);
  return __builtin_shufflevector(lo, hi, 0,1,2,3,4,5,6,7,8,9,10,11,12,13,14,15);
}

// B-matrix 32x16 bf16 built from row-major weights W[N][K] (B[k][n] = W[n][k]):
// lane owns column n = n0+(lane&15), K=[k0..k0+7],[k0+16..k0+23] with k0=kb*32+(lane>>4)*8.
DEV v16bf load_b_frag(const bf16* wbase, int stride, int n0, int kb, int lane) {
  const int n  = n0 + (lane & 15);
  const int k0 = kb * 32 + ((lane >> 4) << 3);
  const bf16* p = wbase + n * stride + k0;
  bf16x8 lo = *(const bf16x8*)p;
  bf16x8 hi = *(const bf16x8*)(p + 16);
  return __builtin_shufflevector(lo, hi, 0,1,2,3,4,5,6,7,8,9,10,11,12,13,14,15);
}

DEV v8f wmma_bf16(v16bf a, v16bf b, v8f c) {
  return __builtin_amdgcn_wmma_f32_16x16x32_bf16(false, a, false, b, (short)0, c,
                                                 false, false);
}

// ---------------------------------------------------------------------------
// Kernel 1: one-time f32 -> bf16 weight conversion.
// wcat rows: [0,128)=W_dt, [128,256)=W_B, [256,384)=W_C  (row-major [N][K=128])
// ---------------------------------------------------------------------------
__global__ void k_convert_weights(const float* __restrict__ Wdt,
                                  const float* __restrict__ Wb,
                                  const float* __restrict__ Wc,
                                  const float* __restrict__ Wout,
                                  bf16* __restrict__ wcat,
                                  bf16* __restrict__ woutb) {
  const int i = blockIdx.x * blockDim.x + threadIdx.x;   // 0..65535
  if (i < 16384)          wcat[i]          = (bf16)Wdt[i];
  else if (i < 32768)     wcat[i]          = (bf16)Wb[i - 16384];
  else if (i < 49152)     wcat[i]          = (bf16)Wc[i - 32768];
  else                    woutb[i - 49152] = (bf16)Wout[i - 49152];
}

// ---------------------------------------------------------------------------
// Kernel 2: per 64-token tile: three WMMA projections + fused epilogue +
// intra-chunk scan (h0 = 0), emitting chunk carries (P = prod a, h_end).
// blockIdx.x = b*NC + c ; 128 threads = 4 waves.
// ---------------------------------------------------------------------------
__global__ __launch_bounds__(128)
void k_proj_scan(const float* __restrict__ x,
                 const float* __restrict__ log_A,
                 const bf16*  __restrict__ wcat,
                 const float* __restrict__ b_dt,
                 float* __restrict__ a_ws, float* __restrict__ u_ws,
                 float* __restrict__ c_ws,
                 float* __restrict__ P_ws, float* __restrict__ He_ws) {
  __shared__ float xf[64 * 128];      // raw f32 x tile (TDM-loaded, 32KB contiguous)
  __shared__ bf16  xb[64 * XPAD];     // bf16 A-operand tile
  __shared__ float dt_l[64 * 128];
  __shared__ float a_l[64 * 128];
  __shared__ float u_l[64 * 128];

  const int  tid  = threadIdx.x;
  const int  lane = tid & 31;
  const int  wave = tid >> 5;
  const long tokbase = (long)blockIdx.x * 64;
  const float* xsrc = x + tokbase * 128;

  // ---- Stage x tile (contiguous 8192 f32) into LDS via the Tensor Data Mover.
  // 1-D descriptor: data_size=4B, tile_dim0=8192, tensor_dim0=8192, type=2.
#if __has_builtin(__builtin_amdgcn_tensor_load_to_lds)
  if (tid < 32) {                                  // one TDM issue per block (wave 0)
    const unsigned lds_off = (unsigned)(uintptr_t)(AS3 char*)&xf[0];
    const unsigned long long ga = (unsigned long long)(uintptr_t)xsrc;
    u32x4 g0 = { 1u,                               // count=1, is_restore=0, gather=0
                 lds_off,                          // lds_addr
                 (unsigned)(ga & 0xffffffffu),     // global_addr[31:0]
                 (unsigned)((ga >> 32) & 0x01ffffffu) | (2u << 30) }; // addr[56:32]|type=2
    i32x8 g1 = { (int)(2u << 16),                  // wg_mask=0, data_size=2 (4B)
                 (int)(8192u << 16),               // tensor_dim0[15:0] @ bits79:48
                 (int)(1u << 16),                  // tensor_dim0 hi=0, tensor_dim1=1
                 (int)(8192u << 16),               // tensor_dim1 hi=0, tile_dim0=8192
                 0,                                // tile_dim1=0 (1-D), tile_dim2=0
                 8192,                             // tensor_dim0_stride lo
                 0, 0 };
    i32x4 z4 = { 0, 0, 0, 0 };
    i32x8 z8 = { 0, 0, 0, 0, 0, 0, 0, 0 };
    __builtin_amdgcn_tensor_load_to_lds(g0, g1, z4, z4, z8, 0);
    __builtin_amdgcn_s_wait_tensorcnt(0);
  }
#else
  for (int i = tid; i < 2048; i += 128)
    async_copy16(xsrc + i * 4, &xf[i * 4]);
  wait_async();
#endif
  __syncthreads();

  for (int i = tid; i < 8192; i += 128) {
    const int r = i >> 7, c = i & 127;
    xb[r * XPAD + c] = (bf16)xf[r * 128 + c];
  }
  __syncthreads();

  const int m0 = wave * 16;
  v16bf afr[4];
#pragma unroll
  for (int kb = 0; kb < 4; ++kb) afr[kb] = load_a_frag(xb, XPAD, m0, kb, lane);

  const int col   = lane & 15;
  const int rbase = (lane >> 4) << 3;

  for (int nt = 0; nt < 24; ++nt) {
    v8f acc = {};
#pragma unroll
    for (int kb = 0; kb < 4; ++kb) {
      v16bf bfr = load_b_frag(wcat, 128, nt * 16, kb, lane);  // weights via L2
      acc = wmma_bf16(afr[kb], bfr, acc);
    }
    const int ncat = nt * 16 + col;
    if (nt < 8) {                       // dt projection -> dt, a
      const int s  = ncat;
      const float bd = b_dt[s];
      const float An = -fminf(fast_exp2(log_A[s] * LOG2E), 10.0f);
#pragma unroll
      for (int r = 0; r < 8; ++r) {
        const int row = m0 + rbase + r;
        const float v  = acc[r] + bd;
        // softplus, saturated at 1: large v -> exp2=inf -> log2=inf -> fmin=1.
        const float e  = fast_exp2(v * LOG2E);
        const float dt = fminf(LN2 * fast_log2(1.0f + e), 1.0f);
        const float a  = fast_exp2(fmaxf(dt * An, -0.5f) * LOG2E);
        dt_l[row * 128 + s] = dt;
        a_l [row * 128 + s] = a;
        a_ws[(tokbase + row) * 128 + s] = a;
      }
    } else if (nt < 16) {               // B projection -> u = (x·W_B)·dt
      const int s = ncat - 128;
#pragma unroll
      for (int r = 0; r < 8; ++r) {
        const int row = m0 + rbase + r;
        const float u = acc[r] * dt_l[row * 128 + s];
        u_l[row * 128 + s] = u;
        u_ws[(tokbase + row) * 128 + s] = u;
      }
    } else {                            // C projection
      const int s = ncat - 256;
#pragma unroll
      for (int r = 0; r < 8; ++r) {
        const int row = m0 + rbase + r;
        c_ws[(tokbase + row) * 128 + s] = acc[r];
      }
    }
  }
  __syncthreads();

  // Intra-chunk scan with h0 = 0; thread owns state s = tid.
  const int s = tid;
  float h = 0.0f, P = 1.0f;
  for (int t = 0; t < 64; ++t) {
    const float a = a_l[t * 128 + s];
    h = fmaf(a, h, u_l[t * 128 + s]);
    P *= a;
  }
  const long bc = blockIdx.x;
  P_ws [bc * 128 + s] = P;
  He_ws[bc * 128 + s] = h;
}

// ---------------------------------------------------------------------------
// Kernel 3: inter-chunk prefix. One block per batch b, thread = state s.
// h0[c+1] = P[c]*h0[c] + h_end[c]
// ---------------------------------------------------------------------------
__global__ __launch_bounds__(128)
void k_chunk_prefix(const float* __restrict__ P_ws,
                    const float* __restrict__ He_ws,
                    float* __restrict__ H0_ws, int NC) {
  const int b = blockIdx.x, s = threadIdx.x;
  float h0 = 0.0f;
  for (int c = 0; c < NC; ++c) {
    const long idx = ((long)b * NC + c) * 128 + s;
    H0_ws[idx] = h0;
    h0 = fmaf(P_ws[idx], h0, He_ws[idx]);
  }
}

// ---------------------------------------------------------------------------
// Kernel 4: re-scan with correct h0, y = sum_s C*h, out = h·W_outᵀ + y*x[:,0].
// ---------------------------------------------------------------------------
__global__ __launch_bounds__(128)
void k_finalize(const float* __restrict__ x,
                const float* __restrict__ a_ws, const float* __restrict__ u_ws,
                const float* __restrict__ c_ws, const float* __restrict__ H0_ws,
                const bf16*  __restrict__ woutb,
                float* __restrict__ out) {
  __shared__ float a_l[64 * 128];
  __shared__ float u_l[64 * 128];
  __shared__ float c_l[64 * 128];
  __shared__ bf16  hb[64 * XPAD];     // bf16 h tile (A operand of output GEMM)
  __shared__ float ch[64 * 129];      // C*h products (pad 129: conflict-free row sums)
  __shared__ float yl[64];
  __shared__ float x0[64];

  const int  tid  = threadIdx.x;
  const int  lane = tid & 31;
  const int  wave = tid >> 5;
  const long tokbase = (long)blockIdx.x * 64;
  const long base    = tokbase * 128;

  for (int i = tid; i < 2048; i += 128) {
    async_copy16(a_ws + base + i * 4, &a_l[i * 4]);
    async_copy16(u_ws + base + i * 4, &u_l[i * 4]);
    async_copy16(c_ws + base + i * 4, &c_l[i * 4]);
  }
  if (tid < 64) x0[tid] = x[(tokbase + tid) * 128];
  wait_async();
  __syncthreads();

  const int s = tid;
  float h = H0_ws[(long)blockIdx.x * 128 + s];
  for (int t = 0; t < 64; ++t) {
    const float a = a_l[t * 128 + s];
    h = fmaf(a, h, u_l[t * 128 + s]);
    hb[t * XPAD + s] = (bf16)h;
    ch[t * 129 + s]  = c_l[t * 128 + s] * h;
  }
  __syncthreads();

  if (tid < 64) {
    float y = 0.0f;
    for (int k = 0; k < 128; ++k) y += ch[tid * 129 + k];
    yl[tid] = y;
  }
  __syncthreads();

  const int m0 = wave * 16;
  v16bf afr[4];
#pragma unroll
  for (int kb = 0; kb < 4; ++kb) afr[kb] = load_a_frag(hb, XPAD, m0, kb, lane);

  const int col   = lane & 15;
  const int rbase = (lane >> 4) << 3;

  for (int nt = 0; nt < 8; ++nt) {
    v8f acc = {};
#pragma unroll
    for (int kb = 0; kb < 4; ++kb) {
      v16bf bfr = load_b_frag(woutb, 128, nt * 16, kb, lane);
      acc = wmma_bf16(afr[kb], bfr, acc);
    }
    const int d = nt * 16 + col;
#pragma unroll
    for (int r = 0; r < 8; ++r) {
      const int row = m0 + rbase + r;
      out[(tokbase + row) * 128 + d] = acc[r] + yl[row] * x0[row];
    }
  }
}

// ---------------------------------------------------------------------------
// Host side
// ---------------------------------------------------------------------------
extern "C" void kernel_launch(void* const* d_in, const int* in_sizes, int n_in,
                              void* d_out, int out_size, void* d_ws, size_t ws_size,
                              hipStream_t stream) {
  (void)n_in; (void)out_size; (void)ws_size;

  const float* x     = (const float*)d_in[0];
  const float* log_A = (const float*)d_in[1];
  const float* W_B   = (const float*)d_in[2];
  const float* W_C   = (const float*)d_in[3];
  const float* W_dt  = (const float*)d_in[4];
  const float* b_dt  = (const float*)d_in[5];
  const float* W_out = (const float*)d_in[6];
  float* out = (float*)d_out;

  const long TOK   = (long)in_sizes[0] / 128;   // B*T tokens
  const int  B     = 8;
  const int  T     = (int)(TOK / B);
  const int  NC    = T / 64;                    // chunks per sequence
  const int  NTILE = (int)(TOK / 64);           // 64-token tiles

  // Workspace carve-up (~102 MB): bf16 weights + a/u/C + chunk carries.
  char* ws = (char*)d_ws;
  bf16* wcat  = (bf16*)ws;  ws += (size_t)49152 * sizeof(bf16);
  bf16* woutb = (bf16*)ws;  ws += (size_t)16384 * sizeof(bf16);
  ws = (char*)(((uintptr_t)ws + 255) & ~(uintptr_t)255);
  float* a_ws  = (float*)ws;  ws += (size_t)TOK * 128 * sizeof(float);
  float* u_ws  = (float*)ws;  ws += (size_t)TOK * 128 * sizeof(float);
  float* c_ws  = (float*)ws;  ws += (size_t)TOK * 128 * sizeof(float);
  float* P_ws  = (float*)ws;  ws += (size_t)B * NC * 128 * sizeof(float);
  float* He_ws = (float*)ws;  ws += (size_t)B * NC * 128 * sizeof(float);
  float* H0_ws = (float*)ws;

  k_convert_weights<<<256, 256, 0, stream>>>(W_dt, W_B, W_C, W_out, wcat, woutb);
  k_proj_scan<<<NTILE, 128, 0, stream>>>(x, log_A, wcat, b_dt,
                                         a_ws, u_ws, c_ws, P_ws, He_ws);
  k_chunk_prefix<<<B, 128, 0, stream>>>(P_ws, He_ws, H0_ws, NC);
  k_finalize<<<NTILE, 128, 0, stream>>>(x, a_ws, u_ws, c_ws, H0_ws, woutb, out);
}
